// DGCN_LSTM_49443663511666
// MI455X (gfx1250) — matrix operand order repaired and checked by reading, not compiled, not run on hardware
//
#include <hip/hip_runtime.h>

// Problem constants (match reference).
#define NN 2048   // agents
#define TT 20     // timesteps
#define KK 32     // neighbors
#define CC 32     // in channels
#define OO 64     // hidden / out channels
#define GG 256    // 4*OO gate width

#define HSTRIDE 72    // bf16 elems per h row in LDS (16B-aligned, bank-spread)
#define GSTRIDE 260   // f32 elems per gate row in LDS (16B-aligned, bank-spread)

typedef __attribute__((ext_vector_type(16))) __bf16 v16bf;
typedef __attribute__((ext_vector_type(8)))  float  v8f;

// Load B-fragment: K = laneHalf*8 + e (e<8), 16 + laneHalf*8 + (e-8) (e>=8).
__device__ __forceinline__ v16bf load_frag16(const __bf16* p) {
  union { v16bf v; uint4 q[2]; } u;
  u.q[0] = *(const uint4*)(p);
  u.q[1] = *(const uint4*)(p + 16);
  return u.v;
}

// Load an A-fragment (16x32 tile) from a bf16 row of 32 channels.
// element e -> channel (e>=8)*16 + laneHalf*8 + (e&7): two 16B chunks.
__device__ __forceinline__ v16bf load_a_frag_bf(const __bf16* row, int lh) {
  union { v16bf v; uint4 q[2]; } u;
  u.q[0] = *(const uint4*)(row + lh * 8);
  u.q[1] = *(const uint4*)(row + 16 + lh * 8);
  return u.v;
}

// Build a bf16 A-fragment of (neighbor_row - center_row) straight from global
// f32 rows; lane loads exactly the 16 channels its fragment elements need.
__device__ __forceinline__ v16bf diff_frag(const float* __restrict__ nrow,
                                           const float* __restrict__ crow,
                                           int lh) {
  float d[16];
  const float4* na = (const float4*)(nrow + lh * 8);
  const float4* ca = (const float4*)(crow + lh * 8);
  const float4* nb = (const float4*)(nrow + 16 + lh * 8);
  const float4* cb = (const float4*)(crow + 16 + lh * 8);
  float4 a0 = na[0], a1 = na[1], b0 = ca[0], b1 = ca[1];
  float4 a2 = nb[0], a3 = nb[1], b2 = cb[0], b3 = cb[1];
  d[0]  = a0.x - b0.x; d[1]  = a0.y - b0.y; d[2]  = a0.z - b0.z; d[3]  = a0.w - b0.w;
  d[4]  = a1.x - b1.x; d[5]  = a1.y - b1.y; d[6]  = a1.z - b1.z; d[7]  = a1.w - b1.w;
  d[8]  = a2.x - b2.x; d[9]  = a2.y - b2.y; d[10] = a2.z - b2.z; d[11] = a2.w - b2.w;
  d[12] = a3.x - b3.x; d[13] = a3.y - b3.y; d[14] = a3.z - b3.z; d[15] = a3.w - b3.w;
  v16bf r;
#pragma unroll
  for (int e = 0; e < 16; ++e) r[e] = (__bf16)d[e];
  return r;
}

__device__ __forceinline__ v8f wmma_bf16(v16bf a, v16bf b, v8f c) {
  return __builtin_amdgcn_wmma_f32_16x16x32_bf16(
      /*neg_a=*/false, a, /*neg_b=*/false, b,
      /*c_mod=*/(short)0, c, /*reuse_a=*/false, /*reuse_b=*/false);
}

__device__ __forceinline__ float sigf(float x) { return 1.0f / (1.0f + expf(-x)); }

// ---------------------------------------------------------------------------
// Kernel 0: one-shot f32 -> bf16 weight conversion into workspace.
// ---------------------------------------------------------------------------
__global__ __launch_bounds__(256) void cvt_weights_kernel(
    const float* __restrict__ cw, const float* __restrict__ wih,
    const float* __restrict__ whh, __bf16* __restrict__ cw_bf,
    __bf16* __restrict__ wih_bf, __bf16* __restrict__ whh_bf) {
  int i = blockIdx.x * 256 + threadIdx.x;
  if (i < OO * CC) cw_bf[i] = (__bf16)cw[i];
  if (i < GG * OO) {
    wih_bf[i] = (__bf16)wih[i];
    whh_bf[i] = (__bf16)whh[i];
  }
}

// ---------------------------------------------------------------------------
// Kernel 1: per (t, agent): gather 32 neighbor rows, diff vs center row,
// [32x32] diff x [32x64] conv_w^T via 8x v_wmma_f32_16x16x32_bf16,
// max over neighbors, +bias, write feat[t][n][o] as bf16 (workspace).
// One wave per (t, n); no LDS: each lane loads exactly its fragment slice.
// ---------------------------------------------------------------------------
__global__ __launch_bounds__(128) void dgcn_conv_max_kernel(
    const float* __restrict__ x, const int* __restrict__ A,
    const __bf16* __restrict__ cw_bf, const float* __restrict__ conv_b,
    __bf16* __restrict__ feat) {
  const int w    = threadIdx.x >> 5;
  const int lane = threadIdx.x & 31;
  const int lh   = lane >> 4;
  const int lm   = lane & 15;
  const int gw   = blockIdx.x * 4 + w;   // t*NN + n
  const int t    = gw >> 11;             // / 2048
  const int n    = gw & (NN - 1);

  // A-fragment rows: this lane's M row is neighbor lm (tile 0) / lm+16 (tile 1).
  const size_t abase = (size_t)t * NN * KK + (size_t)n * KK;
  const int idx0 = A[abase + lm];
  const int idx1 = A[abase + 16 + lm];
  const float* rc = x + (size_t)n * TT * CC + (size_t)t * CC;
  v16bf afr0 = diff_frag(x + (size_t)idx0 * TT * CC + (size_t)t * CC, rc, lh);
  v16bf afr1 = diff_frag(x + (size_t)idx1 * TT * CC + (size_t)t * CC, rc, lh);

  v8f acc[2][4];
#pragma unroll
  for (int ot = 0; ot < 4; ++ot) {
    const int o = ot * 16 + lm;  // this lane's output column
    v16bf bfr = load_frag16(cw_bf + (size_t)o * CC + lh * 8);
    v8f z = {};
    acc[0][ot] = wmma_bf16(afr0, bfr, z);
    acc[1][ot] = wmma_bf16(afr1, bfr, z);
  }

  // Max over 32 neighbors: 8 in-lane rows per tile, two tiles, then cross
  // lane-half combine (rows 0-7 vs 8-15).
#pragma unroll
  for (int ot = 0; ot < 4; ++ot) {
    float m = -3.4e38f;
#pragma unroll
    for (int r = 0; r < 8; ++r)
      m = fmaxf(m, fmaxf(acc[0][ot][r], acc[1][ot][r]));
    m = fmaxf(m, __shfl_xor(m, 16, 32));
    if (lane < 16) {
      const int o = ot * 16 + lm;
      feat[(size_t)t * NN * OO + (size_t)n * OO + o] = (__bf16)(m + conv_b[o]);
    }
  }
}

// ---------------------------------------------------------------------------
// Kernel 2: LSTM. Block = 256 threads (8 waves) owns 16 agents for all T
// steps. Wave w keeps the bf16 B-fragments of its 2 gate tiles (w_ih & w_hh,
// two K=32 chunks each) in registers for the whole time loop; feat and h are
// consumed as packed-bf16 A-fragments (b128 loads, zero conversion VALU).
// ---------------------------------------------------------------------------
__global__ __launch_bounds__(256) void dgcn_lstm_kernel(
    const __bf16* __restrict__ feat,
    const __bf16* __restrict__ wih_bf, const __bf16* __restrict__ whh_bf,
    const float* __restrict__ b_ih, const float* __restrict__ b_hh,
    float* __restrict__ out) {
  __shared__ __align__(16) __bf16 hbuf[16 * HSTRIDE];  // [agent][channel] bf16
  __shared__ float gbuf[16 * GSTRIDE];                 // [agent][gate] f32

  const int tid  = threadIdx.x;
  const int w    = tid >> 5;
  const int lane = tid & 31;
  const int lh   = lane >> 4;
  const int lm   = lane & 15;
  const int agent0 = blockIdx.x * 16;

  // Hoisted weight B-fragments + per-column bias for tiles j = 2w, 2w+1.
  v16bf bih[2][2], bhh[2][2];
  float bias[2];
#pragma unroll
  for (int ti = 0; ti < 2; ++ti) {
    const int col = (w * 2 + ti) * 16 + lm;  // gate index (0..255)
    bias[ti] = b_ih[col] + b_hh[col];
#pragma unroll
    for (int kc = 0; kc < 2; ++kc) {
      bih[ti][kc] = load_frag16(wih_bf + (size_t)col * OO + kc * 32 + lh * 8);
      bhh[ti][kc] = load_frag16(whh_bf + (size_t)col * OO + kc * 32 + lh * 8);
    }
  }

  // h0 = 0 (bf16 LDS), c0/h kept in registers by the elementwise mapping:
  // thread tid owns agent a_e = tid>>4, channels u0..u0+3.
  for (int i = tid; i < 16 * HSTRIDE; i += 256) hbuf[i] = (__bf16)0.0f;
  float4 creg = {0.f, 0.f, 0.f, 0.f};
  float4 hreg = {0.f, 0.f, 0.f, 0.f};
  const int a_e = tid >> 4;
  const int u0  = (tid & 15) * 4;
  __syncthreads();

  for (int t = 0; t < TT; ++t) {
    // A-fragments (M = agent rows): feat_t from global (bf16), h from LDS.
    const __bf16* frow = feat + (size_t)t * NN * OO + (size_t)(agent0 + lm) * OO;
    v16bf af0 = load_a_frag_bf(frow, lh);
    v16bf af1 = load_a_frag_bf(frow + 32, lh);
    v16bf ah0 = load_a_frag_bf(&hbuf[lm * HSTRIDE], lh);
    v16bf ah1 = load_a_frag_bf(&hbuf[lm * HSTRIDE + 32], lh);

#pragma unroll
    for (int ti = 0; ti < 2; ++ti) {
      const int j = w * 2 + ti;
      v8f acc = {bias[ti], bias[ti], bias[ti], bias[ti],
                 bias[ti], bias[ti], bias[ti], bias[ti]};
      acc = wmma_bf16(af0, bih[ti][0], acc);
      acc = wmma_bf16(af1, bih[ti][1], acc);
      acc = wmma_bf16(ah0, bhh[ti][0], acc);
      acc = wmma_bf16(ah1, bhh[ti][1], acc);
      // C layout: lane column = j*16+lm, VGPR r = agent row lh*8+r.
#pragma unroll
      for (int r = 0; r < 8; ++r)
        gbuf[(lh * 8 + r) * GSTRIDE + j * 16 + lm] = acc[r];
    }
    __syncthreads();

    // Elementwise LSTM update (f32). jnp.split order: i, f, g, o.
    const float* gr = &gbuf[a_e * GSTRIDE];
    float4 ig = *(const float4*)(gr + u0);
    float4 fg = *(const float4*)(gr + 64 + u0);
    float4 gg = *(const float4*)(gr + 128 + u0);
    float4 og = *(const float4*)(gr + 192 + u0);
    creg.x = sigf(fg.x) * creg.x + sigf(ig.x) * tanhf(gg.x);
    creg.y = sigf(fg.y) * creg.y + sigf(ig.y) * tanhf(gg.y);
    creg.z = sigf(fg.z) * creg.z + sigf(ig.z) * tanhf(gg.z);
    creg.w = sigf(fg.w) * creg.w + sigf(ig.w) * tanhf(gg.w);
    hreg.x = sigf(og.x) * tanhf(creg.x);
    hreg.y = sigf(og.y) * tanhf(creg.y);
    hreg.z = sigf(og.z) * tanhf(creg.z);
    hreg.w = sigf(og.w) * tanhf(creg.w);
    __bf16* hw = &hbuf[a_e * HSTRIDE + u0];
    hw[0] = (__bf16)hreg.x; hw[1] = (__bf16)hreg.y;
    hw[2] = (__bf16)hreg.z; hw[3] = (__bf16)hreg.w;
    *(float4*)&out[(size_t)(agent0 + a_e) * TT * OO + (size_t)t * OO + u0] = hreg;
    __syncthreads();
  }

  // Final states: out layout = [N*T*O] ++ hN[N*O] ++ cN[N*O].
  const size_t baseH = (size_t)NN * TT * OO;
  const size_t baseC = baseH + (size_t)NN * OO;
  *(float4*)&out[baseH + (size_t)(agent0 + a_e) * OO + u0] = hreg;
  *(float4*)&out[baseC + (size_t)(agent0 + a_e) * OO + u0] = creg;
}

// ---------------------------------------------------------------------------
extern "C" void kernel_launch(void* const* d_in, const int* in_sizes, int n_in,
                              void* d_out, int out_size, void* d_ws, size_t ws_size,
                              hipStream_t stream) {
  (void)in_sizes; (void)n_in; (void)out_size; (void)ws_size;
  const float* x      = (const float*)d_in[0];
  const int*   A      = (const int*)  d_in[1];
  const float* conv_w = (const float*)d_in[2];
  const float* conv_b = (const float*)d_in[3];
  const float* w_ih   = (const float*)d_in[4];
  const float* w_hh   = (const float*)d_in[5];
  const float* b_ih   = (const float*)d_in[6];
  const float* b_hh   = (const float*)d_in[7];
  float* out = (float*)d_out;

  // Workspace layout (bf16): feat [T,N,O], then conv_w, w_ih, w_hh copies.
  char* ws = (char*)d_ws;
  __bf16* feat_bf = (__bf16*)ws;                               // 5,242,880 B
  __bf16* cw_bf   = (__bf16*)(ws + (size_t)TT * NN * OO * 2);  // 4,096 B
  __bf16* wih_bf  = (__bf16*)((char*)cw_bf + OO * CC * 2);     // 32,768 B
  __bf16* whh_bf  = (__bf16*)((char*)wih_bf + GG * OO * 2);    // 32,768 B

  hipLaunchKernelGGL(cvt_weights_kernel, dim3((GG * OO) / 256), dim3(256), 0,
                     stream, conv_w, w_ih, w_hh, cw_bf, wih_bf, whh_bf);

  dim3 b1(128), g1((TT * NN) / 4);   // one wave per (t, agent)
  hipLaunchKernelGGL(dgcn_conv_max_kernel, g1, b1, 0, stream,
                     x, A, cw_bf, conv_b, feat_bf);

  dim3 b2(256), g2(NN / 16);         // 16 agents per block, 8 waves split gates
  hipLaunchKernelGGL(dgcn_lstm_kernel, g2, b2, 0, stream,
                     feat_bf, wih_bf, whh_bf, b_ih, b_hh, out);
}
